// Deform_Conv_21414706938551
// MI455X (gfx1250) — matrix operand order repaired and checked by reading, not compiled, not run on hardware
//
#include <hip/hip_runtime.h>

typedef __attribute__((ext_vector_type(16))) _Float16 v16h;
typedef __attribute__((ext_vector_type(8)))  float    v8f;
// 2-float pair with 4B alignment: lets backend use b64 loads on unaligned-capable HW
typedef __attribute__((ext_vector_type(2), aligned(4))) float f2a;

#define Bn   8
#define Cc   64
#define Hh   96
#define Ww   96
#define Oo   64
#define HW   (Hh * Ww)          // 9216
#define KTOT (Cc * 9)           // 576
#define NOFF (Bn * 18 * HW)     // offset elements (5.3 MB as f32)

// ---------------- Kernel 1: offset-predicting 3x3 conv (64 -> 18 ch) -------
__global__ __launch_bounds__(256)
void off_conv_kernel(const float* __restrict__ x, const float* __restrict__ w_off,
                     const float* __restrict__ b_off, float* __restrict__ offs) {
    int idx = blockIdx.x * 256 + threadIdx.x;
    if (idx >= NOFF) return;
    int wo = idx % Ww;
    int t  = idx / Ww;
    int ho = t % Hh; t /= Hh;
    int ch = t % 18; t /= 18;
    int b  = t;

    float sum = b_off[ch];
    const float* xb = x + (size_t)b * Cc * HW;
    const float* wb = w_off + (size_t)ch * Cc * 9;
    for (int c = 0; c < Cc; ++c) {
        const float* xp = xb + (size_t)c * HW;
        const float* wp = wb + c * 9;
        #pragma unroll
        for (int ky = 0; ky < 3; ++ky) {
            int y = ho - 1 + ky;
            if ((unsigned)y >= (unsigned)Hh) continue;
            #pragma unroll
            for (int kx = 0; kx < 3; ++kx) {
                int xx = wo - 1 + kx;
                if ((unsigned)xx >= (unsigned)Ww) continue;
                sum += xp[y * Ww + xx] * wp[ky * 3 + kx];
            }
        }
    }
    offs[idx] = sum;
}

// ------- Kernel 1b: convert weights to f16, permuted tap-major -------------
// Contraction index kIdx = tap*64 + c ; wf16[o*576 + tap*64 + c] = w_dcn[o,c,tap]
__global__ __launch_bounds__(256)
void wcvt_kernel(const float* __restrict__ w_dcn, _Float16* __restrict__ wf16) {
    int idx = blockIdx.x * 256 + threadIdx.x;
    if (idx >= Oo * KTOT) return;
    int o   = idx / KTOT;
    int r   = idx - o * KTOT;
    int tap = r >> 6;
    int c   = r & 63;
    wf16[idx] = (_Float16)w_dcn[(size_t)o * KTOT + c * 9 + tap];
}

// ---------------- Kernel 2: deformable gather + WMMA GEMM ------------------
// 1 wave per 16-position tile (one image-row segment); wave does all 64 out-ch.
// GEMM: M=73728, N=64, K=576 (tap-major), f16 in / f32 acc.
// Bilinear sample per (m,tap) uses 2 paired row loads; horizontal weights are
// remapped onto a safe in-bounds pair [bx, bx+1] in the metadata phase.
__global__ __launch_bounds__(256)
void dcn_wmma_kernel(const float* __restrict__ x, const _Float16* __restrict__ wf16,
                     const float* __restrict__ offs, float* __restrict__ out) {
    __shared__ int2   s_addr[8][16 * 9];   // {top-row base, bottom-row base}
    __shared__ float4 s_wgt [8][16 * 9];   // {vy0*e0, vy0*e1, vy1*e0, vy1*e1}

    const int wave = threadIdx.x >> 5;
    const int lane = threadIdx.x & 31;
    const int hi   = lane >> 4;
    const int col  = lane & 15;

    const int tileM  = blockIdx.x * 8 + wave;
    const int m_base = tileM * 16;

    // Tile-uniform: 16 positions sit in one row of one image (16 | Ww | HW)
    const int bb    = m_base / HW;
    const int rmb   = m_base - bb * HW;
    const int ho    = rmb / Ww;
    const int wbase = rmb - ho * Ww;
    const int wo    = wbase + col;

    // ---- Phase 1: per-(row, tap) paired-load metadata ----
    if (lane < 16) {
        const float* offp = offs + (size_t)bb * 18 * HW + (size_t)ho * Ww + wo;
        #pragma unroll
        for (int k = 0; k < 9; ++k) {
            int ky = k / 3, kx = k - ky * 3;
            float oy = offp[(2 * k) * HW];
            float ox = offp[(2 * k + 1) * HW];
            float py = oy + (float)(ho - 1 + ky);
            float px = ox + (float)(wo - 1 + kx);
            float fy = floorf(py), fx = floorf(px);
            int   y0 = (int)fy,    x0 = (int)fx;
            float wy = py - fy,    wx = px - fx;
            int   y1 = y0 + 1,     x1 = x0 + 1;

            // validity-folded separable weights
            float hx0  = ((x0 >= 0) && (x0 < Ww)) ? (1.f - wx) : 0.f;
            float hx1  = ((x1 >= 0) && (x1 < Ww)) ? wx          : 0.f;
            float vy0w = ((y0 >= 0) && (y0 < Hh)) ? (1.f - wy) : 0.f;
            float vy1w = ((y1 >= 0) && (y1 < Hh)) ? wy          : 0.f;

            // remap horizontal weights onto safe pair [bx, bx+1]
            int   bx = min(max(x0, 0), Ww - 2);
            float e0 = ((x0 == bx)     ? hx0 : 0.f) + ((x1 == bx)     ? hx1 : 0.f);
            float e1 = ((x0 == bx + 1) ? hx0 : 0.f) + ((x1 == bx + 1) ? hx1 : 0.f);

            int cy0 = min(max(y0, 0), Hh - 1);
            int cy1 = min(max(y1, 0), Hh - 1);

            s_addr[wave][col * 9 + k] = int2{cy0 * Ww + bx, cy1 * Ww + bx};
            s_wgt [wave][col * 9 + k] = float4{vy0w * e0, vy0w * e1,
                                               vy1w * e0, vy1w * e1};
        }
    }
    __syncthreads();

    const float* xb = x + (size_t)bb * Cc * HW;
    const int koffA = hi ? 16 : 0;    // 16b A 16x32 K-layout (contiguous k = 16h + s)
    const int koffB = hi ? 16 : 0;    // 16b B 32x16 K-layout

    v8f acc[4] = {v8f{}, v8f{}, v8f{}, v8f{}};

    for (int ks = 0; ks < KTOT; ks += 32) {
        // tap-major K: all kk in [ks, ks+32) share tap = ks>>6
        const int tap = ks >> 6;
        const int c0  = (ks & 63) + koffA;
        int2   aa = s_addr[wave][col * 9 + tap];
        float4 wf = s_wgt [wave][col * 9 + tap];

        // ---- A fragment: 16 bilinear samples, 2 paired loads each ----
        v16h a;
        #pragma unroll
        for (int i = 0; i < 16; ++i) {
            int c = c0 + i;
            const float* pl = xb + (size_t)c * HW;
            f2a tp = *(const f2a*)(pl + aa.x);
            f2a bt = *(const f2a*)(pl + aa.y);
            float v = wf.x * tp.x + wf.y * tp.y + wf.z * bt.x + wf.w * bt.y;
            a[i] = (_Float16)v;
        }
        // ---- 4 N-tiles: preconverted f16 B fragment (32B aligned), WMMA ----
        #pragma unroll
        for (int nt = 0; nt < 4; ++nt) {
            v16h bm = *(const v16h*)(wf16 + (size_t)(nt * 16 + col) * KTOT
                                     + ks + koffB);
            acc[nt] = __builtin_amdgcn_wmma_f32_16x16x32_f16(
                false, a, false, bm, (short)0, acc[nt], false, false);
        }
    }

    // ---- store D (VGPR j -> row j + 8*hi, col = lane&15) ----
    const size_t outrow = (size_t)bb * Oo * HW + (size_t)ho * Ww + wbase;
    #pragma unroll
    for (int j = 0; j < 8; ++j) {
        int row = j + (hi ? 8 : 0);
        #pragma unroll
        for (int nt = 0; nt < 4; ++nt) {
            out[outrow + (size_t)(nt * 16 + col) * HW + row] = acc[nt][j];
        }
    }
}

extern "C" void kernel_launch(void* const* d_in, const int* in_sizes, int n_in,
                              void* d_out, int out_size, void* d_ws, size_t ws_size,
                              hipStream_t stream) {
    const float* x     = (const float*)d_in[0];
    const float* w_off = (const float*)d_in[1];
    const float* b_off = (const float*)d_in[2];
    const float* w_dcn = (const float*)d_in[3];
    float* out  = (float*)d_out;

    float*    offs = (float*)d_ws;                                 // NOFF floats
    _Float16* wf16 = (_Float16*)((char*)d_ws + (size_t)NOFF * 4);  // 72 KB, 32B-aligned

    off_conv_kernel<<<(NOFF + 255) / 256, 256, 0, stream>>>(x, w_off, b_off, offs);
    wcvt_kernel<<<(Oo * KTOT + 255) / 256, 256, 0, stream>>>(w_dcn, wf16);

    // 73728 positions / 16 rows = 4608 M-tiles; 8 waves/block -> 576 blocks
    dcn_wmma_kernel<<<576, 256, 0, stream>>>(x, wf16, offs, out);
}
